// InformerAttentionLayer_23261542875385
// MI455X (gfx1250) — matrix operand order repaired and checked by reading, not compiled, not run on hardware
//
#include <hip/hip_runtime.h>

// ---------------------------------------------------------------------------
// Informer ProbSparse attention layer for MI455X (gfx1250, wave32, WMMA+TDM).
//
// Roofline: ~36 GFLOP total, ~95% in four 4096x1024x1024 GEMMs; ~250 MB of
// HBM traffic (~11us @ 23.3 TB/s) -> matrix-compute bound. Strategy:
//  * One elementwise pass splits fp32 operands into bf16 hi/lo planes
//    (x, Wq/Wk/Wv/Wo, attn_out). Near-fp32 accuracy via 3-term WMMA:
//    Ah*Bh + Ah*Bl + Al*Bh accumulated in f32 (error ~2^-16).
//  * GEMM tiles are moved global->LDS by the Tensor Data Mover
//    (tensor_load_to_lds, double-buffered, TENSORcnt-synchronized), with
//    TDM row padding (128B rows + 16B pad -> 144B pitch) to spread LDS banks.
//  * Compute: V_WMMA_F32_16X16X32_BF16, 64x64 block tile, 8 waves,
//    12 WMMAs per wave per 64-wide K chunk.
// Workspace requirement: ~128 MB.
// ---------------------------------------------------------------------------

#define DEV static __device__ __forceinline__

typedef __attribute__((ext_vector_type(16))) __bf16          v16bf;
typedef __attribute__((ext_vector_type(16))) unsigned short  v16us;
typedef __attribute__((ext_vector_type(8)))  unsigned short  v8us;
typedef __attribute__((ext_vector_type(8)))  float           v8f;
typedef __attribute__((ext_vector_type(4)))  unsigned int    v4u;
typedef __attribute__((ext_vector_type(8)))  int             v8i_;
typedef __attribute__((ext_vector_type(4)))  int             v4i_;

constexpr int B_  = 2;
constexpr int H_  = 16;
constexpr int L_  = 2048;
constexpr int DM  = 1024;
constexpr int DH  = 64;
constexpr int SK  = 35;          // sample_k == top_k == 5*int(log(2048))
constexpr int MROWS = B_ * L_;   // 4096
constexpr int BH    = B_ * H_;   // 32
constexpr int KC    = 64;        // GEMM K chunk
constexpr int PITCH = 72;        // LDS row pitch in shorts (128B data + 16B TDM pad)

DEV unsigned short f2bf(float f) {              // round-to-nearest-even
  unsigned u = __float_as_uint(f);
  u += 0x7FFFu + ((u >> 16) & 1u);
  return (unsigned short)(u >> 16);
}
DEV float bf2f(unsigned short h) { return __uint_as_float(((unsigned)h) << 16); }

// Assemble a v16bf WMMA fragment from two 16-byte aligned LDS runs
// (lowers to 2x ds_load_b128).
DEV v16bf mkfrag(const unsigned short* p0, const unsigned short* p1) {
  v8us a = *(const v8us*)p0;
  v8us b = *(const v8us*)p1;
  v16us r;
#pragma unroll
  for (int i = 0; i < 8; ++i) { r[i] = a[i]; r[i + 8] = b[i]; }
  return __builtin_bit_cast(v16bf, r);
}

DEV v8f wmma_bf16(v16bf a, v16bf b, v8f c) {
  return __builtin_amdgcn_wmma_f32_16x16x32_bf16(false, a, false, b, (short)0, c,
                                                 false, false);
}

// ---------------------------------------------------------------------------
// TDM: DMA one 64-row x 64-element bf16 tile (row pitch DM elements in global)
// into LDS at lds_addr, inserting a 16B pad after every 128B row.
// D# group0/group1 per CDNA5 ISA 8.3/8.4; groups 2/3 unused (2D tile).
// Toolchain uses the 6-arg builtin form: (v4u, v8i, v4i, v4i, v8i, cpol).
// ---------------------------------------------------------------------------
DEV void tdm_load_tile64x64(const unsigned short* src, unsigned lds_addr) {
  unsigned long long ga = (unsigned long long)(size_t)src;
  v4u g0 = { 1u,                                    // count=1, user mode
             lds_addr,                              // LDS byte address
             (unsigned)(ga & 0xFFFFFFFFull),        // global_addr[31:0]
             (unsigned)((ga >> 32) & 0x1FFFFFFull)  // global_addr[56:32]
                 | (2u << 30) };                    // type=2 (image)
  v8i_ g1 = {
    (int)((1u << 16)       // data_size=1 -> 2 bytes
        | (1u << 20)       // pad_enable
        | (4u << 22)       // pad_interval code 4 -> every 32 DWORDs (128B)
        | (3u << 25)),     // pad_amount  code 3 -> 4 DWORDs (16B)
    (int)(((unsigned)DM & 0xFFFFu) << 16),            // tensor_dim0 low16
    (int)(((unsigned)DM >> 16) | (0xFFFFu << 16)),    // dim0 hi | tensor_dim1=0xFFFF
    (int)(64u << 16),                                 // dim1 hi=0 | tile_dim0=64
    (int)(64u),                                       // tile_dim1=64, tile_dim2=0
    (int)(DM),                                        // tensor_dim0_stride=1024
    0, 0 };                                           // stride0 hi / stride1
  v4i_ z4 = {};
  v8i_ z8 = {};
  __builtin_amdgcn_tensor_load_to_lds(g0, g1, z4, z4, z8, 0);
}

// ---------------------------------------------------------------------------
// C = A(4096x1024) @ W^T(1024x1024) + bias, operands pre-split into bf16
// hi/lo planes. TDM double-buffered tile staging, bf16x2-split WMMA compute.
// headmajor=1: out[(b*H+h)*L*DH + l*DH + d];  headmajor=0: out[i*DM + j]
// ---------------------------------------------------------------------------
__global__ __launch_bounds__(256)
void gemm2_kernel(const unsigned short* __restrict__ Ahp, const unsigned short* __restrict__ Alp,
                  const unsigned short* __restrict__ Whp, const unsigned short* __restrict__ Wlp,
                  const float* __restrict__ bias, float* __restrict__ out, int headmajor) {
  // [buffer][plane: 0=Ah 1=Al 2=Wh 3=Wl][64 rows * PITCH]
  __shared__ __align__(16) unsigned short lds[2][4][64 * PITCH];

  const int tid  = threadIdx.x;
  const int lane = tid & 31;
  const int wave = tid >> 5;       // 0..7
  const int mw   = wave >> 2;      // 0..1 -> 32-row slice
  const int nw   = wave & 3;       // 0..3 -> 16-col slice
  const int l15  = lane & 15;
  const int hi   = lane >> 4;      // lane half (ISA 16-bit operand layout)

  const int m0 = blockIdx.x * 64;
  const int n0 = blockIdx.y * 64;

  v8f acc0 = {};
  v8f acc1 = {};

  const unsigned short* srcA[2] = { Ahp, Alp };
  const unsigned short* srcW[2] = { Whp, Wlp };

  // prime buffer 0
  if (wave == 0) {
#pragma unroll
    for (int p = 0; p < 2; ++p) {
      tdm_load_tile64x64(srcA[p] + (size_t)m0 * DM, (unsigned)(size_t)&lds[0][p][0]);
      tdm_load_tile64x64(srcW[p] + (size_t)n0 * DM, (unsigned)(size_t)&lds[0][2 + p][0]);
    }
  }

  for (int kc = 0; kc < DM / KC; ++kc) {
    const int buf = kc & 1;
    if (wave == 0) __builtin_amdgcn_s_wait_tensorcnt(0);  // chunk kc landed
    __syncthreads();

    if (kc + 1 < DM / KC && wave == 0) {                  // overlap DMA of kc+1
      const int k1 = (kc + 1) * KC;
#pragma unroll
      for (int p = 0; p < 2; ++p) {
        tdm_load_tile64x64(srcA[p] + (size_t)m0 * DM + k1,
                           (unsigned)(size_t)&lds[1 - buf][p][0]);
        tdm_load_tile64x64(srcW[p] + (size_t)n0 * DM + k1,
                           (unsigned)(size_t)&lds[1 - buf][2 + p][0]);
      }
    }

    const unsigned short* LAh = lds[buf][0];
    const unsigned short* LAl = lds[buf][1];
    const unsigned short* LWh = lds[buf][2];
    const unsigned short* LWl = lds[buf][3];
    const int nr = nw * 16 + l15;

#pragma unroll
    for (int s = 0; s < 2; ++s) {                 // two 32-wide K steps
      const int kof = s * 32;
      // B fragment: b[0..7]=W[n][kb..], b[8..15]=W[n][kb+16..], kb = kof+hi*8
      const int kb = kof + hi * 8;
      v16bf Bh = mkfrag(&LWh[nr * PITCH + kb], &LWh[nr * PITCH + kb + 16]);
      v16bf Bl = mkfrag(&LWl[nr * PITCH + kb], &LWl[nr * PITCH + kb + 16]);
      // A fragment: a[0..7]=A[m][ba..], a[8..15]=A[m][ba+16..], ba = kof+hi*8
      const int ba = kof + hi * 8;
      {
        const int mr = mw * 32 + l15;
        v16bf Af = mkfrag(&LAh[mr * PITCH + ba], &LAh[mr * PITCH + ba + 16]);
        v16bf Ag = mkfrag(&LAl[mr * PITCH + ba], &LAl[mr * PITCH + ba + 16]);
        acc0 = wmma_bf16(Af, Bh, acc0);
        acc0 = wmma_bf16(Af, Bl, acc0);
        acc0 = wmma_bf16(Ag, Bh, acc0);
      }
      {
        const int mr = mw * 32 + 16 + l15;
        v16bf Af = mkfrag(&LAh[mr * PITCH + ba], &LAh[mr * PITCH + ba + 16]);
        v16bf Ag = mkfrag(&LAl[mr * PITCH + ba], &LAl[mr * PITCH + ba + 16]);
        acc1 = wmma_bf16(Af, Bh, acc1);
        acc1 = wmma_bf16(Af, Bl, acc1);
        acc1 = wmma_bf16(Ag, Bh, acc1);
      }
    }
    __syncthreads();   // all waves done with lds[buf] before it is re-filled
  }

  // C/D layout: VGPR r -> M = r + hi*8, N = lane15
#pragma unroll
  for (int sub = 0; sub < 2; ++sub) {
    v8f acc = sub ? acc1 : acc0;
#pragma unroll
    for (int r = 0; r < 8; ++r) {
      int i = m0 + mw * 32 + sub * 16 + r + hi * 8;
      int j = n0 + nw * 16 + l15;
      float v = acc[r] + bias[j];
      if (headmajor) {
        int b = i >> 11, l = i & (L_ - 1);
        int h = j >> 6,  d = j & 63;
        out[(((size_t)(b * H_ + h)) * L_ + l) * DH + d] = v;
      } else {
        out[(size_t)i * DM + j] = v;
      }
    }
  }
}

// fp32 -> bf16 hi/lo split planes (done once per tensor)
__global__ __launch_bounds__(256)
void split_kernel(const float* __restrict__ src, unsigned short* __restrict__ hi,
                  unsigned short* __restrict__ lo, int n) {
  int i = blockIdx.x * 256 + threadIdx.x;
  if (i >= n) return;
  float v = src[i];
  unsigned short h = f2bf(v);
  hi[i] = h;
  lo[i] = f2bf(v - bf2f(h));
}

// ---------------------------------------------------------------------------
// M[b,h,l] = max_s(Q[l].K[idx[l,s]]) - mean_s(...)
// ---------------------------------------------------------------------------
__global__ __launch_bounds__(256)
void sampled_metric_kernel(const float* __restrict__ Qh, const float* __restrict__ Kh,
                           const int* __restrict__ idx, float* __restrict__ Mout) {
  int gid = blockIdx.x * 256 + threadIdx.x;     // bh*L + l
  if (gid >= BH * L_) return;
  int bh = gid >> 11;
  int l  = gid & (L_ - 1);
  const float4* q = (const float4*)(Qh + (size_t)gid * DH);
  float mx = -3.4e38f, sum = 0.f;
  for (int s = 0; s < SK; ++s) {
    int j = idx[l * SK + s];
    const float4* kr = (const float4*)(Kh + ((size_t)bh * L_ + j) * DH);
    float d = 0.f;
#pragma unroll
    for (int i = 0; i < 16; ++i) {
      float4 a = q[i], b = kr[i];
      d += a.x * b.x + a.y * b.y + a.z * b.z + a.w * b.w;
    }
    mx = fmaxf(mx, d);
    sum += d;
  }
  Mout[gid] = mx - sum * (1.f / SK);
}

// ---------------------------------------------------------------------------
// top-35 of M per (b,h): iterative argmax over 2048 with shared reduction
// ---------------------------------------------------------------------------
__global__ __launch_bounds__(256)
void topk_kernel(const float* __restrict__ Mbuf, int* __restrict__ topk) {
  __shared__ float vals[L_];
  __shared__ float rv[256];
  __shared__ int   ri[256];
  int bh = blockIdx.x, tid = threadIdx.x;
  for (int l = tid; l < L_; l += 256) vals[l] = Mbuf[bh * L_ + l];
  __syncthreads();
  for (int k = 0; k < SK; ++k) {
    float bv = -3.4e38f; int bi = 0x7fffffff;
    for (int l = tid; l < L_; l += 256) {
      float v = vals[l];
      if (v > bv) { bv = v; bi = l; }
    }
    rv[tid] = bv; ri[tid] = bi;
    __syncthreads();
    for (int off = 128; off > 0; off >>= 1) {
      if (tid < off) {
        float v2 = rv[tid + off]; int i2 = ri[tid + off];
        if (v2 > rv[tid] || (v2 == rv[tid] && i2 < ri[tid])) { rv[tid] = v2; ri[tid] = i2; }
      }
      __syncthreads();
    }
    if (tid == 0) { topk[bh * SK + k] = ri[0]; vals[ri[0]] = -3.4e38f; }
    __syncthreads();
  }
}

// ---------------------------------------------------------------------------
// attn[b,h,kq,:] = softmax( Q[top_idx[kq]] . K[l] / 8 )  over l
// ---------------------------------------------------------------------------
__global__ __launch_bounds__(256)
void attn_softmax_kernel(const float* __restrict__ Qh, const float* __restrict__ Kh,
                         const int* __restrict__ topk, float* __restrict__ attnw) {
  int bh = blockIdx.x / SK, kq = blockIdx.x % SK, tid = threadIdx.x;
  __shared__ float q[DH];
  __shared__ float sc[L_];
  __shared__ float red[256];
  int qi = topk[bh * SK + kq];
  if (tid < DH) q[tid] = Qh[((size_t)bh * L_ + qi) * DH + tid];
  __syncthreads();
  const float4* q4 = (const float4*)q;
  for (int l = tid; l < L_; l += 256) {
    const float4* kr = (const float4*)(Kh + ((size_t)bh * L_ + l) * DH);
    float d = 0.f;
#pragma unroll
    for (int i = 0; i < 16; ++i) {
      float4 a = q4[i], b = kr[i];
      d += a.x * b.x + a.y * b.y + a.z * b.z + a.w * b.w;
    }
    sc[l] = d * 0.125f;   // 1/sqrt(64)
  }
  __syncthreads();
  float mx = -3.4e38f;
  for (int l = tid; l < L_; l += 256) mx = fmaxf(mx, sc[l]);
  red[tid] = mx; __syncthreads();
  for (int off = 128; off > 0; off >>= 1) {
    if (tid < off) red[tid] = fmaxf(red[tid], red[tid + off]);
    __syncthreads();
  }
  mx = red[0];
  __syncthreads();
  float s = 0.f;
  for (int l = tid; l < L_; l += 256) { float e = __expf(sc[l] - mx); sc[l] = e; s += e; }
  red[tid] = s; __syncthreads();
  for (int off = 128; off > 0; off >>= 1) {
    if (tid < off) red[tid] += red[tid + off];
    __syncthreads();
  }
  float inv = 1.f / red[0];
  float* arow = attnw + (size_t)blockIdx.x * L_;
  for (int l = tid; l < L_; l += 256) arow[l] = sc[l] * inv;
}

// ctx[bh,kq,d] = sum_l attn[bh,kq,l] * V[bh,l,d]
__global__ __launch_bounds__(64)
void context_kernel(const float* __restrict__ attnw, const float* __restrict__ Vh,
                    float* __restrict__ ctx) {
  int bk = blockIdx.x;        // bh*SK + kq
  int bh = bk / SK;
  int d  = threadIdx.x;
  const float* arow = attnw + (size_t)bk * L_;
  const float* vb   = Vh + (size_t)bh * L_ * DH + d;
  float acc = 0.f;
  for (int l = 0; l < L_; ++l) acc += arow[l] * vb[(size_t)l * DH];
  ctx[(size_t)bk * DH + d] = acc;
}

__global__ __launch_bounds__(64)
void vmean_kernel(const float* __restrict__ Vh, float* __restrict__ vmean) {
  int bh = blockIdx.x, d = threadIdx.x;
  const float* vb = Vh + (size_t)bh * L_ * DH + d;
  float s = 0.f;
  for (int l = 0; l < L_; ++l) s += vb[(size_t)l * DH];
  vmean[bh * DH + d] = s * (1.f / L_);
}

// attn_out[b,l,h*64+d] = vmean[b,h,d]
__global__ __launch_bounds__(256)
void broadcast_kernel(const float* __restrict__ vmean, float* __restrict__ attn_out) {
  size_t gid = (size_t)blockIdx.x * 256 + threadIdx.x;  // < B*L*DM
  int    j   = (int)(gid & (DM - 1));
  size_t bl  = gid >> 10;                               // b*L + l
  int    b   = (int)(bl >> 11);
  int    h   = j >> 6, d = j & 63;
  attn_out[gid] = vmean[(b * H_ + h) * DH + d];
}

// attn_out[b, top_idx[bh,kq], h*64+d] = ctx[bh,kq,d]
__global__ __launch_bounds__(256)
void scatter_kernel(const float* __restrict__ ctx, const int* __restrict__ topk,
                    float* __restrict__ attn_out) {
  int gid = blockIdx.x * 256 + threadIdx.x;   // < BH*SK*DH
  if (gid >= BH * SK * DH) return;
  int d  = gid & 63;
  int t  = gid >> 6;                          // bh*SK + kq
  int bh = t / SK;
  int b = bh >> 4, h = bh & 15;
  int li = topk[t];
  attn_out[((size_t)b * L_ + li) * DM + h * DH + d] = ctx[gid];
}

// ---------------------------------------------------------------------------
extern "C" void kernel_launch(void* const* d_in, const int* in_sizes, int n_in,
                              void* d_out, int out_size, void* d_ws, size_t ws_size,
                              hipStream_t stream) {
  (void)in_sizes; (void)n_in; (void)out_size; (void)ws_size;
  const float* x   = (const float*)d_in[0];
  const float* Wq  = (const float*)d_in[1];
  const float* bq  = (const float*)d_in[2];
  const float* Wk  = (const float*)d_in[3];
  const float* bk  = (const float*)d_in[4];
  const float* Wv  = (const float*)d_in[5];
  const float* bv  = (const float*)d_in[6];
  const float* Wo  = (const float*)d_in[7];
  const float* bo  = (const float*)d_in[8];
  const int*   idx = (const int*)d_in[9];
  float* out = (float*)d_out;

  // ---- workspace carve-up (floats first, then int, then bf16 planes) ----
  float* ws = (float*)d_ws;
  const size_t TEN = (size_t)BH * L_ * DH;    // 4,194,304 (= B*L*DM)
  float* Qh       = ws; ws += TEN;
  float* Kh       = ws; ws += TEN;
  float* Vh       = ws; ws += TEN;
  float* attn_out = ws; ws += TEN;
  float* Mbuf     = ws; ws += (size_t)BH * L_;
  float* attnw    = ws; ws += (size_t)BH * SK * L_;
  float* ctx      = ws; ws += (size_t)BH * SK * DH;
  float* vmean    = ws; ws += (size_t)BH * DH;
  int*   topk     = (int*)ws;
  unsigned short* us = (unsigned short*)(topk + ((BH * SK + 63) & ~63));
  const size_t NX = (size_t)MROWS * DM;       // 4,194,304
  const size_t NW = (size_t)DM * DM;          // 1,048,576
  unsigned short* xh  = us; us += NX;
  unsigned short* xl  = us; us += NX;
  unsigned short* aoh = us; us += NX;
  unsigned short* aol = us; us += NX;
  unsigned short* wqh = us; us += NW;  unsigned short* wql = us; us += NW;
  unsigned short* wkh = us; us += NW;  unsigned short* wkl = us; us += NW;
  unsigned short* wvh = us; us += NW;  unsigned short* wvl = us; us += NW;
  unsigned short* woh = us; us += NW;  unsigned short* wol = us; us += NW;

  dim3 gg(MROWS / 64, DM / 64);   // (64, 16)

  // one-time bf16 hi/lo splits
  split_kernel<<<(int)(NX / 256), 256, 0, stream>>>(x,  xh,  xl,  (int)NX);
  split_kernel<<<(int)(NW / 256), 256, 0, stream>>>(Wq, wqh, wql, (int)NW);
  split_kernel<<<(int)(NW / 256), 256, 0, stream>>>(Wk, wkh, wkl, (int)NW);
  split_kernel<<<(int)(NW / 256), 256, 0, stream>>>(Wv, wvh, wvl, (int)NW);
  split_kernel<<<(int)(NW / 256), 256, 0, stream>>>(Wo, woh, wol, (int)NW);

  // Q/K/V projections (TDM-staged WMMA GEMMs, head-major output)
  gemm2_kernel<<<gg, 256, 0, stream>>>(xh, xl, wqh, wql, bq, Qh, 1);
  gemm2_kernel<<<gg, 256, 0, stream>>>(xh, xl, wkh, wkl, bk, Kh, 1);
  gemm2_kernel<<<gg, 256, 0, stream>>>(xh, xl, wvh, wvl, bv, Vh, 1);

  // ProbSparse attention
  sampled_metric_kernel<<<(BH * L_) / 256, 256, 0, stream>>>(Qh, Kh, idx, Mbuf);
  topk_kernel<<<BH, 256, 0, stream>>>(Mbuf, topk);
  attn_softmax_kernel<<<BH * SK, 256, 0, stream>>>(Qh, Kh, topk, attnw);
  context_kernel<<<BH * SK, 64, 0, stream>>>(attnw, Vh, ctx);
  vmean_kernel<<<BH, 64, 0, stream>>>(Vh, vmean);
  broadcast_kernel<<<(int)((size_t)B_ * L_ * DM / 256), 256, 0, stream>>>(vmean, attn_out);
  scatter_kernel<<<(BH * SK * DH + 255) / 256, 256, 0, stream>>>(ctx, topk, attn_out);

  // output projection
  split_kernel<<<(int)(NX / 256), 256, 0, stream>>>(attn_out, aoh, aol, (int)NX);
  gemm2_kernel<<<gg, 256, 0, stream>>>(aoh, aol, woh, wol, bo, out, 0);
}
